// So3kratesLayer_83915071029331
// MI455X (gfx1250) — matrix operand order repaired and checked by reading, not compiled, not run on hardware
//
#include <hip/hip_runtime.h>
#include <hip/hip_bf16.h>

// ---------------------------------------------------------------------------
// So3krates layer for MI455X (gfx1250), wave32 + WMMA bf16.
// ---------------------------------------------------------------------------

typedef __attribute__((ext_vector_type(16))) __bf16 v16bf;
typedef __attribute__((ext_vector_type(8)))  float  v8f;

union BF16Frag { v16bf v; __bf16 e[16]; unsigned short s[16]; uint4 q[2]; };
union F32x8    { v8f   v; float f[8]; };

// native bf16 convert (RTNE in hardware); clang lowers to v_cvt_*bf16_f32
static __device__ __forceinline__ __bf16 tobf(float x) { return (__bf16)x; }

static __device__ __forceinline__ float silu_f(float x) {
  return x / (1.f + __expf(-x));
}

// problem sizes
#define NN  16000
#define EE  512000
#define FF  128
#define SHD 15
// merged MLP dims: x[64] -> h[448] -> y[144]  (cols 0..127 = w, 128..130 = wg)
#define K1  64
#define H1  448
#define O2  144
#define KT1 2
#define NT1 28
#define KT2 14
#define NT2 9

// workspace byte offsets
#define OFF_WCAT1 0u                       // 2*28*512 bf16  = 57344 B
#define OFF_WCAT2 57344u                   // 14*9*512 bf16  = 129024 B
#define OFF_WPROJ 186368u                  // 4 * 4*8*512 bf16 = 131072 B
#define OFF_BIAS1 317440u                  // 448 f32
#define OFF_BIAS2 319232u                  // 144 f32
#define OFF_Q     327680u
#define OFF_K     (OFF_Q    + 8192000u)
#define OFF_QG    (OFF_K    + 8192000u)
#define OFF_KG    (OFF_QG   + 8192000u)
#define OFF_ACCN  (OFF_KG   + 8192000u)
#define OFF_ACCC  (OFF_ACCN + 8192000u)    // end = 42247680 B

static __device__ __forceinline__ v8f wmma_bf16(const BF16Frag& a, const BF16Frag& b, v8f c) {
  return __builtin_amdgcn_wmma_f32_16x16x32_bf16(false, a.v, false, b.v, (short)0, c, false, false);
}

// ---------------------------------------------------------------------------
// 0) Pack weights into bf16 B-fragment register images + merged biases.
//    B frag element layout: idx = frag*512 + lane*16 + e ;
//    n = nt*16 + (lane&15), k = kt*32 + (e>>3)*16 + (lane>>4)*8 + (e&7).
// ---------------------------------------------------------------------------
__global__ void pack_weights(
    const float* __restrict__ fbradW0, const float* __restrict__ fbradB0,
    const float* __restrict__ fbradW1, const float* __restrict__ fbradB1,
    const float* __restrict__ fbsphW0, const float* __restrict__ fbsphB0,
    const float* __restrict__ fbsphW1, const float* __restrict__ fbsphB1,
    const float* __restrict__ gbradW0, const float* __restrict__ gbradB0,
    const float* __restrict__ gbradW1, const float* __restrict__ gbradB1,
    const float* __restrict__ gbsphW0, const float* __restrict__ gbsphB0,
    const float* __restrict__ gbsphW1, const float* __restrict__ gbsphB1,
    const float* __restrict__ fbq, const float* __restrict__ fbk,
    const float* __restrict__ gbq, const float* __restrict__ gbk,
    char* __restrict__ ws)
{
  __bf16* wcat1 = (__bf16*)(ws + OFF_WCAT1);
  __bf16* wcat2 = (__bf16*)(ws + OFF_WCAT2);
  __bf16* wproj = (__bf16*)(ws + OFF_WPROJ);
  float* bias1 = (float*)(ws + OFF_BIAS1);
  float* bias2 = (float*)(ws + OFF_BIAS2);
  const int n1 = KT1 * NT1 * 512;     // 28672
  const int n2 = KT2 * NT2 * 512;     // 64512
  const int np = 4 * 4 * 8 * 512;     // 65536
  const int total = n1 + n2 + np + H1 + O2;
  for (int idx = blockIdx.x * blockDim.x + threadIdx.x; idx < total;
       idx += gridDim.x * blockDim.x) {
    if (idx < n1) {                                   // Wcat1 [64 x 448]
      int frag = idx >> 9, r = idx & 511;
      int kt = frag / NT1, nt = frag % NT1;
      int lane = r >> 4, e = r & 15;
      int n = nt * 16 + (lane & 15);
      int k = kt * 32 + (e >> 3) * 16 + (lane >> 4) * 8 + (e & 7);
      float v = 0.f;
      if (k < 32) {
        if (n < 128)                     v = fbradW0[k * 128 + n];
        else if (n >= 256 && n < 384)    v = gbradW0[k * 128 + (n - 256)];
      } else {
        int kk = k - 32;
        if (kk < 3) {
          if (n >= 128 && n < 256)       v = fbsphW0[kk * 128 + (n - 128)];
          else if (n >= 384)             v = gbsphW0[kk * 64 + (n - 384)];
        }
      }
      wcat1[idx] = tobf(v);
    } else if (idx < n1 + n2) {                       // Wcat2 [448 x 144]
      int i2 = idx - n1;
      int frag = i2 >> 9, r = i2 & 511;
      int kt = frag / NT2, nt = frag % NT2;
      int lane = r >> 4, e = r & 15;
      int n = nt * 16 + (lane & 15);
      int k = kt * 32 + (e >> 3) * 16 + (lane >> 4) * 8 + (e & 7);
      float v = 0.f;
      if (n < 128) {
        if (k < 128)        v = fbradW1[k * 128 + n];
        else if (k < 256)   v = fbsphW1[(k - 128) * 128 + n];
      } else {
        int nn = n - 128;
        if (nn < 3) {
          if (k >= 256 && k < 384)  v = gbradW1[(k - 256) * 3 + nn];
          else if (k >= 384)        v = gbsphW1[(k - 384) * 3 + nn];
        }
      }
      wcat2[i2] = tobf(v);
    } else if (idx < n1 + n2 + np) {                  // four 128x128 proj mats
      int i3 = idx - n1 - n2;
      int which = i3 / 16384;
      int i4 = i3 % 16384;
      int frag = i4 >> 9, r = i4 & 511;
      int kt = frag >> 3, nt = frag & 7;
      int lane = r >> 4, e = r & 15;
      int n = nt * 16 + (lane & 15);
      int k = kt * 32 + (e >> 3) * 16 + (lane >> 4) * 8 + (e & 7);
      const float* W = (which == 0) ? fbq : (which == 1) ? fbk
                     : (which == 2) ? gbq : gbk;
      wproj[i3] = tobf(W[k * 128 + n]);
    } else if (idx < n1 + n2 + np + H1) {             // bias1[448]
      int c = idx - n1 - n2 - np;
      bias1[c] = (c < 128) ? fbradB0[c] : (c < 256) ? fbsphB0[c - 128]
               : (c < 384) ? gbradB0[c - 256] : gbsphB0[c - 384];
    } else {                                          // bias2[144]
      int c = idx - n1 - n2 - np - H1;
      float v = 0.f;
      if (c < 128)      v = fbradB1[c] + fbsphB1[c];
      else if (c < 131) v = gbradB1[c - 128] + gbsphB1[c - 128];
      bias2[c] = v;
    }
  }
}

// ---------------------------------------------------------------------------
// 1) init accumulators: acc_node = node_feats, acc_chi = chi
// ---------------------------------------------------------------------------
__global__ void init_acc(const float* __restrict__ nf, const float* __restrict__ chi,
                         float* __restrict__ accn, float* __restrict__ accc)
{
  int idx = blockIdx.x * blockDim.x + threadIdx.x;
  int stride = gridDim.x * blockDim.x;
  for (int i = idx; i < NN * FF; i += stride)  accn[i] = nf[i];
  for (int i = idx; i < NN * SHD; i += stride) accc[i] = chi[i];
}

// ---------------------------------------------------------------------------
// 2) node projections: out = nf @ W  (W in bf16 B-frags), blockIdx.y picks
//    {fb_q, fb_k, gb_q, gb_k}. One 16-row tile per wave, 8 waves/block.
// ---------------------------------------------------------------------------
__global__ __launch_bounds__(256) void proj_kernel(
    const float* __restrict__ nf, const char* __restrict__ ws, char* __restrict__ wso)
{
  const int lane = threadIdx.x & 31, wid = threadIdx.x >> 5;
  const int hf = lane >> 4, nn = lane & 15;
  const int r0 = (blockIdx.x * 8 + wid) * 16;
  const __bf16* wfrag = (const __bf16*)(ws + OFF_WPROJ) + (size_t)blockIdx.y * 16384;
  float* out = (float*)(wso + OFF_Q) + (size_t)blockIdx.y * (NN * FF);

  BF16Frag A[4];
  #pragma unroll
  for (int kt = 0; kt < 4; ++kt) {
    const float* base = nf + (size_t)(r0 + nn) * FF + kt * 32 + hf * 8;
    float4 f0 = *(const float4*)(base);
    float4 f1 = *(const float4*)(base + 4);
    float4 f2 = *(const float4*)(base + 16);
    float4 f3 = *(const float4*)(base + 20);
    float t[16] = {f0.x,f0.y,f0.z,f0.w, f1.x,f1.y,f1.z,f1.w,
                   f2.x,f2.y,f2.z,f2.w, f3.x,f3.y,f3.z,f3.w};
    #pragma unroll
    for (int j = 0; j < 16; ++j) A[kt].e[j] = tobf(t[j]);
  }
  #pragma unroll
  for (int nt = 0; nt < 8; ++nt) {
    F32x8 acc;
    #pragma unroll
    for (int j = 0; j < 8; ++j) acc.f[j] = 0.f;
    #pragma unroll
    for (int kt = 0; kt < 4; ++kt) {
      BF16Frag B;
      const uint4* bp = (const uint4*)(wfrag + (kt * 8 + nt) * 512) + lane * 2;
      B.q[0] = bp[0]; B.q[1] = bp[1];
      acc.v = wmma_bf16(A[kt], B, acc.v);
    }
    const int col = nt * 16 + nn;
    #pragma unroll
    for (int j = 0; j < 8; ++j)
      out[(size_t)(r0 + j + 8 * hf) * FF + col] = acc.f[j];
  }
}

// ---------------------------------------------------------------------------
// 3) fused edge kernel: chi_scalar -> merged MLP (WMMA) -> attention + scatter
//    4 waves/block, 16 edges/wave (E = 8000 * 64).
// ---------------------------------------------------------------------------
__global__ __launch_bounds__(128) void edge_kernel(
    const float* __restrict__ edge_feats, const float* __restrict__ chi,
    const int* __restrict__ senders, const int* __restrict__ receivers,
    const float* __restrict__ cutoffs, const float* __restrict__ edge_sh,
    const float* __restrict__ node_feats,
    const char* __restrict__ ws, char* __restrict__ wsm)
{
  __shared__ __bf16 h_lds[4][16 * H1];   // 57344 B
  __shared__ int   s_lds[4][16];
  __shared__ int   r_lds[4][16];
  __shared__ float co_lds[4][16];
  __shared__ float wg_lds[4][16 * 3];

  const __bf16* wcat1 = (const __bf16*)(ws + OFF_WCAT1);
  const __bf16* wcat2 = (const __bf16*)(ws + OFF_WCAT2);
  const float* bias1 = (const float*)(ws + OFF_BIAS1);
  const float* bias2 = (const float*)(ws + OFF_BIAS2);
  const float* qbuf  = (const float*)(ws + OFF_Q);
  const float* kbuf  = (const float*)(ws + OFF_K);
  const float* qgbuf = (const float*)(ws + OFF_QG);
  const float* kgbuf = (const float*)(ws + OFF_KG);
  float* accn = (float*)(wsm + OFF_ACCN);
  float* accc = (float*)(wsm + OFF_ACCC);

  const int lane = threadIdx.x & 31, wid = threadIdx.x >> 5;
  const int hf = lane >> 4, nn = lane & 15;
  const int e0 = (blockIdx.x * 4 + wid) * 16;
  const int e  = e0 + nn;

  const int   sE = senders[e], rE = receivers[e];
  const float coE = cutoffs[e];
  if (hf == 0) { s_lds[wid][nn] = sE; r_lds[wid][nn] = rE; co_lds[wid][nn] = coE; }

  // per-irrep squared norms of chi difference
  float cs0 = 0.f, cs1 = 0.f, cs2 = 0.f;
  {
    const float* cS = chi + (size_t)sE * SHD;
    const float* cR = chi + (size_t)rE * SHD;
    #pragma unroll
    for (int t = 0; t < SHD; ++t) {
      float d = cS[t] - cR[t];
      if (t < 3) cs0 += d * d; else if (t < 8) cs1 += d * d; else cs2 += d * d;
    }
  }

  // A-fragments for x = [edge_feats(32) | cs(3) pad(29)]
  BF16Frag A0, A1;
  {
    const float* base = edge_feats + (size_t)e * 32 + hf * 8;
    float4 f0 = *(const float4*)(base);
    float4 f1 = *(const float4*)(base + 4);
    float4 f2 = *(const float4*)(base + 16);
    float4 f3 = *(const float4*)(base + 20);
    float t[16] = {f0.x,f0.y,f0.z,f0.w, f1.x,f1.y,f1.z,f1.w,
                   f2.x,f2.y,f2.z,f2.w, f3.x,f3.y,f3.z,f3.w};
    #pragma unroll
    for (int j = 0; j < 16; ++j) A0.e[j] = tobf(t[j]);
  }
  #pragma unroll
  for (int j = 0; j < 16; ++j) A1.s[j] = 0;
  if (hf == 0) { A1.e[0] = tobf(cs0); A1.e[1] = tobf(cs1); A1.e[2] = tobf(cs2); }

  // ---- GEMM1: h = silu(x @ Wcat1 + b1) -> LDS (bf16, row-major [16][448]) ----
  for (int nt = 0; nt < NT1; ++nt) {
    const int col = nt * 16 + nn;
    F32x8 acc;
    float b = bias1[col];
    #pragma unroll
    for (int j = 0; j < 8; ++j) acc.f[j] = b;
    BF16Frag B;
    const uint4* bp0 = (const uint4*)(wcat1 + (0 * NT1 + nt) * 512) + lane * 2;
    B.q[0] = bp0[0]; B.q[1] = bp0[1];
    acc.v = wmma_bf16(A0, B, acc.v);
    const uint4* bp1 = (const uint4*)(wcat1 + (1 * NT1 + nt) * 512) + lane * 2;
    B.q[0] = bp1[0]; B.q[1] = bp1[1];
    acc.v = wmma_bf16(A1, B, acc.v);
    #pragma unroll
    for (int j = 0; j < 8; ++j)
      h_lds[wid][(j + 8 * hf) * H1 + col] = tobf(silu_f(acc.f[j]));
  }
  __syncthreads();

  // hoist all GEMM2 A-fragments into registers (14 x 8 VGPRs)
  BF16Frag hA[KT2];
  #pragma unroll
  for (int kt = 0; kt < KT2; ++kt) {
    const __bf16* hb = &h_lds[wid][nn * H1 + kt * 32 + hf * 8];
    hA[kt].q[0] = *(const uint4*)(hb);
    hA[kt].q[1] = *(const uint4*)(hb + 16);
  }

  // ---- GEMM2 heads: w = h @ Wcat2 + b2, attention on D-fragments ----
  const float ATT_SC = 0.005524271728019903f;     // 1/sqrt(32) / 32
  for (int head = 0; head < 4; ++head) {
    const int ntA = 2 * head, ntB = ntA + 1;
    F32x8 accA, accB;
    {
      float bA = bias2[ntA * 16 + nn], bB = bias2[ntB * 16 + nn];
      #pragma unroll
      for (int j = 0; j < 8; ++j) { accA.f[j] = bA; accB.f[j] = bB; }
    }
    #pragma unroll
    for (int kt = 0; kt < KT2; ++kt) {
      BF16Frag B;
      const uint4* bpA = (const uint4*)(wcat2 + (kt * NT2 + ntA) * 512) + lane * 2;
      B.q[0] = bpA[0]; B.q[1] = bpA[1];
      accA.v = wmma_bf16(hA[kt], B, accA.v);
      const uint4* bpB = (const uint4*)(wcat2 + (kt * NT2 + ntB) * 512) + lane * 2;
      B.q[0] = bpB[0]; B.q[1] = bpB[1];
      accB.v = wmma_bf16(hA[kt], B, accB.v);
    }
    const int colA = ntA * 16 + nn, colB = colA + 16;
    float p[8]; int rj[8], sj[8];
    #pragma unroll
    for (int j = 0; j < 8; ++j) {
      int row = j + 8 * hf;
      int r_ = r_lds[wid][row], s_ = s_lds[wid][row];
      rj[j] = r_; sj[j] = s_;
      p[j] = qbuf[(size_t)r_ * FF + colA] * accA.f[j] * kbuf[(size_t)s_ * FF + colA]
           + qbuf[(size_t)r_ * FF + colB] * accB.f[j] * kbuf[(size_t)s_ * FF + colB];
    }
    #pragma unroll
    for (int mask = 1; mask < 16; mask <<= 1) {
      #pragma unroll
      for (int j = 0; j < 8; ++j) p[j] += __shfl_xor(p[j], mask, 32);
    }
    #pragma unroll
    for (int j = 0; j < 8; ++j) {
      int row = j + 8 * hf;
      float alpha = p[j] * co_lds[wid][row] * ATT_SC;
      atomicAdd(&accn[(size_t)rj[j] * FF + colA],
                alpha * node_feats[(size_t)sj[j] * FF + colA]);
      atomicAdd(&accn[(size_t)rj[j] * FF + colB],
                alpha * node_feats[(size_t)sj[j] * FF + colB]);
    }
  }

  // ---- GEMM2 geometric ntile (cols 128..130 = wg) ----
  {
    F32x8 accG;
    float bG = bias2[128 + nn];
    #pragma unroll
    for (int j = 0; j < 8; ++j) accG.f[j] = bG;
    #pragma unroll
    for (int kt = 0; kt < KT2; ++kt) {
      BF16Frag B;
      const uint4* bp = (const uint4*)(wcat2 + (kt * NT2 + 8) * 512) + lane * 2;
      B.q[0] = bp[0]; B.q[1] = bp[1];
      accG.v = wmma_bf16(hA[kt], B, accG.v);
    }
    if (nn < 3) {
      #pragma unroll
      for (int j = 0; j < 8; ++j) wg_lds[wid][(j + 8 * hf) * 3 + nn] = accG.f[j];
    }
  }

  // a = <qg[r], kg[s]> / sqrt(128): each lane sums 64 cols, pair-combine halves
  float aa = 0.f;
  {
    const float* qr = qgbuf + (size_t)rE * FF + hf * 64;
    const float* kk = kgbuf + (size_t)sE * FF + hf * 64;
    #pragma unroll
    for (int c = 0; c < 64; ++c) aa += qr[c] * kk[c];
  }
  aa += __shfl_xor(aa, 16, 32);
  __syncthreads();
  if (hf == 0) {
    float sc = aa * 0.08838834764831845f * coE * 0.03125f;   // /sqrt(128)/32
    float al0 = wg_lds[wid][nn * 3 + 0] * sc;
    float al1 = wg_lds[wid][nn * 3 + 1] * sc;
    float al2 = wg_lds[wid][nn * 3 + 2] * sc;
    const float* esh = edge_sh + (size_t)e * SHD;
    float* dst = accc + (size_t)rE * SHD;
    #pragma unroll
    for (int t = 0; t < SHD; ++t) {
      float al = (t < 3) ? al0 : (t < 8) ? al1 : al2;
      atomicAdd(&dst[t], al * esh[t]);
    }
  }
}

// ---------------------------------------------------------------------------
// 4) interaction block: y = [nf_mid | irrep_norm2(chi_mid)] @ int_w + int_b
// ---------------------------------------------------------------------------
__global__ __launch_bounds__(128) void interaction_kernel(
    const char* __restrict__ wsm, const float* __restrict__ int_w,
    const float* __restrict__ int_b, float* __restrict__ out)
{
  const float* accn = (const float*)(wsm + OFF_ACCN);
  const float* accc = (const float*)(wsm + OFF_ACCC);
  __shared__ float x[131];
  __shared__ float xchi[15];
  __shared__ float gate[3];
  const int node = blockIdx.x;
  const int t = threadIdx.x;
  float nfm = accn[(size_t)node * FF + t];
  x[t] = nfm;
  if (t < 15) xchi[t] = accc[(size_t)node * SHD + t];
  __syncthreads();
  if (t < 3) {
    int s0 = (t == 0) ? 0 : (t == 1) ? 3 : 8;
    int s1 = (t == 0) ? 3 : (t == 1) ? 8 : 15;
    float a = 0.f;
    for (int i = s0; i < s1; ++i) a += xchi[i] * xchi[i];
    x[128 + t] = a;
  }
  __syncthreads();
  float y = int_b[t];
  for (int i = 0; i < 131; ++i) y += x[i] * int_w[i * 131 + t];
  out[(size_t)node * FF + t] = nfm + y;
  if (t < 3) {
    float yg = int_b[128 + t];
    for (int i = 0; i < 131; ++i) yg += x[i] * int_w[i * 131 + 128 + t];
    gate[t] = yg;
  }
  __syncthreads();
  if (t < 15) {
    float g = (t < 3) ? gate[0] : (t < 8) ? gate[1] : gate[2];
    out[(size_t)NN * FF + (size_t)node * SHD + t] = xchi[t] * (1.f + g);
  }
}

// ---------------------------------------------------------------------------
extern "C" void kernel_launch(void* const* d_in, const int* in_sizes, int n_in,
                              void* d_out, int out_size, void* d_ws, size_t ws_size,
                              hipStream_t stream)
{
  (void)in_sizes; (void)n_in; (void)out_size; (void)ws_size;
  const float* node_feats = (const float*)d_in[0];
  const float* chi        = (const float*)d_in[1];
  const float* edge_feats = (const float*)d_in[2];
  const float* edge_sh    = (const float*)d_in[3];
  const float* cutoffs    = (const float*)d_in[4];
  const int*   senders    = (const int*)d_in[5];
  const int*   receivers  = (const int*)d_in[6];
  // params flattened in dict insertion order
  const float* fbradW0 = (const float*)d_in[7];
  const float* fbradB0 = (const float*)d_in[8];
  const float* fbradW1 = (const float*)d_in[9];
  const float* fbradB1 = (const float*)d_in[10];
  const float* fbsphW0 = (const float*)d_in[11];
  const float* fbsphB0 = (const float*)d_in[12];
  const float* fbsphW1 = (const float*)d_in[13];
  const float* fbsphB1 = (const float*)d_in[14];
  const float* gbradW0 = (const float*)d_in[15];
  const float* gbradB0 = (const float*)d_in[16];
  const float* gbradW1 = (const float*)d_in[17];
  const float* gbradB1 = (const float*)d_in[18];
  const float* gbsphW0 = (const float*)d_in[19];
  const float* gbsphB0 = (const float*)d_in[20];
  const float* gbsphW1 = (const float*)d_in[21];
  const float* gbsphB1 = (const float*)d_in[22];
  const float* fbq     = (const float*)d_in[23];
  const float* fbk     = (const float*)d_in[24];
  const float* gbq     = (const float*)d_in[25];
  const float* gbk     = (const float*)d_in[26];
  const float* int_w   = (const float*)d_in[27];
  const float* int_b   = (const float*)d_in[28];

  char* ws = (char*)d_ws;
  float* out = (float*)d_out;

  pack_weights<<<312, 256, 0, stream>>>(
      fbradW0, fbradB0, fbradW1, fbradB1, fbsphW0, fbsphB0, fbsphW1, fbsphB1,
      gbradW0, gbradB0, gbradW1, gbradB1, gbsphW0, gbsphB0, gbsphW1, gbsphB1,
      fbq, fbk, gbq, gbk, ws);

  init_acc<<<512, 256, 0, stream>>>(node_feats, chi,
                                    (float*)(ws + OFF_ACCN), (float*)(ws + OFF_ACCC));

  dim3 pg(125, 4);  // 1000 row tiles of 16, 8 tiles/block; y = which weight
  proj_kernel<<<pg, 256, 0, stream>>>(node_feats, ws, ws);

  edge_kernel<<<EE / 64, 128, 0, stream>>>(edge_feats, chi, senders, receivers,
                                           cutoffs, edge_sh, node_feats, ws, ws);

  interaction_kernel<<<NN, 128, 0, stream>>>(ws, int_w, int_b, out);
}
